// MultiHeadedAttention_34033320854026
// MI455X (gfx1250) — matrix operand order repaired and checked by reading, not compiled, not run on hardware
//
#include <hip/hip_runtime.h>
#include <stdint.h>

// ---------------------------------------------------------------------------
// MI455X (gfx1250) multi-head attention, bf16 WMMA with f32 accumulate.
// B=2, N=2048, D=1024, H=16, DK=64.  Compute-bound => everything through
// v_wmma_f32_16x16x32_bf16.
// Round 3: fold softmax scale * log2(e) into the Q projection (softmax runs in
// base-2, bare v_exp_f32), and cap the key-stream loop unroll to keep the
// attention kernel under the 256-VGPR direct-addressing window.
// ---------------------------------------------------------------------------

typedef __attribute__((ext_vector_type(16))) __bf16       v16bf;
typedef __attribute__((ext_vector_type(8)))  float        v8f;
typedef __attribute__((ext_vector_type(4)))  unsigned int u32x4;

#define MB   2
#define MN   2048
#define MD   1024
#define MH   16
#define MDK  64
// DK^-0.5 * log2(e): folded into Q so S from WMMA is already in log2 units.
#define MQSCALE 0.18033688011112042f

union Frag16 { v16bf v; u32x4 q[2]; };

__device__ __forceinline__ uint16_t f2bf(float f) {
  uint32_t u = __float_as_uint(f);
  u += 0x7fffu + ((u >> 16) & 1u);       // round-to-nearest-even
  return (uint16_t)(u >> 16);
}

// --------------------------- fp32 -> bf16 ----------------------------------
__global__ void cvt_f32_bf16(const float* __restrict__ in,
                             uint16_t* __restrict__ out, int n) {
  int i = blockIdx.x * blockDim.x + threadIdx.x;
  if (i < n) out[i] = f2bf(in[i]);
}

// --------------------------- projection GEMM -------------------------------
// C = X(bf16)[Mx1024] * W^T(bf16) + bias, W given row-major [out,in]
// (== B^T row-major: B-frag loads are two 16B runs along k per lane).
// Wave tile 32x64: each W fragment feeds 2 WMMAs, each A fragment feeds 4.
// MODE 0: -> Qh [b,h,n,dk] bf16, scaled by DK^-0.5*log2(e)
// MODE 1: -> Kh [b,h,n,dk] bf16
// MODE 2: -> Vt [b,h,dk,n] bf16
// MODE 3: -> out [b,n,D] fp32
template <int MODE>
__global__ __launch_bounds__(128) void proj_wmma(
    const uint16_t* __restrict__ X, const uint16_t* __restrict__ W,
    const float* __restrict__ bias, void* __restrict__ dst) {
  const int lane = threadIdx.x & 31;
  const int wave = threadIdx.x >> 5;
  const int lr   = lane & 15;
  const int hf   = lane >> 4;
  const int r0   = blockIdx.y * 128 + wave * 32;  // 32 rows per wave
  const int c0   = blockIdx.x * 64;               // 64 cols per wave (4 tiles)

  v8f acc[2][4] = {{{}, {}, {}, {}}, {{}, {}, {}, {}}};
  const uint16_t* xrow0 = X + (size_t)(r0 + lr) * MD;
  const uint16_t* xrow1 = X + (size_t)(r0 + 16 + lr) * MD;

  for (int k = 0; k < MD; k += 32) {
    Frag16 a[2];  // A 16x32: lane=row, e<8 -> k=hf*8+e ; e>=8 -> 16+hf*8+(e-8)
    a[0].q[0] = *(const u32x4*)(xrow0 + k + hf * 8);
    a[0].q[1] = *(const u32x4*)(xrow0 + k + 16 + hf * 8);
    a[1].q[0] = *(const u32x4*)(xrow1 + k + hf * 8);
    a[1].q[1] = *(const u32x4*)(xrow1 + k + 16 + hf * 8);
#pragma unroll
    for (int t = 0; t < 4; ++t) {
      Frag16 bf;  // B 32x16: lane=col, e<8 -> k=hf*8+e ; e>=8 -> 16+hf*8+(e-8)
      const uint16_t* wrow = W + (size_t)(c0 + t * 16 + lr) * MD + k + hf * 8;
      bf.q[0] = *(const u32x4*)(wrow);
      bf.q[1] = *(const u32x4*)(wrow + 16);
#pragma unroll
      for (int r = 0; r < 2; ++r)
        acc[r][t] = __builtin_amdgcn_wmma_f32_16x16x32_bf16(
            false, a[r].v, false, bf.v, (short)0, acc[r][t], false, false);
    }
  }

#pragma unroll
  for (int t = 0; t < 4; ++t) {
    const int col = c0 + t * 16 + lr;
    const float bv = bias[col];
#pragma unroll
    for (int r = 0; r < 2; ++r)
#pragma unroll
      for (int v = 0; v < 8; ++v) {         // C/D: lane=col, row = v + 8*hf
        float val = acc[r][t][v] + bv;
        if (MODE == 0) val *= MQSCALE;      // fold softmax scale into Q
        const int row = r0 + r * 16 + v + 8 * hf;  // token index in [0, B*N)
        if (MODE == 3) {
          ((float*)dst)[(size_t)row * MD + col] = val;
        } else {
          const int b  = row >> 11;           // / N
          const int n  = row & (MN - 1);
          const int h  = col >> 6;            // / DK
          const int dk = col & (MDK - 1);
          const uint16_t bfv = f2bf(val);
          if (MODE == 2)
            ((uint16_t*)dst)[((size_t)(b * MH + h) * MDK + dk) * MN + n] = bfv;
          else
            ((uint16_t*)dst)[((size_t)(b * MH + h) * MN + n) * MDK + dk] = bfv;
        }
      }
  }
}

// --------------------------- flash attention -------------------------------
// One wave per 32-query block per (b,h): each streamed K/V fragment feeds two
// WMMAs.  Streams keys in chunks of 32; online softmax in base-2 f32 (scale
// pre-folded into Q); QK^T and P*V both on v_wmma bf16; prefetches next chunk.
__global__ __launch_bounds__(32) void attn_wmma(
    const uint16_t* __restrict__ Qh,  // [b,h,n,dk] (pre-scaled)
    const uint16_t* __restrict__ Kh,  // [b,h,n,dk]
    const uint16_t* __restrict__ Vt,  // [b,h,dk,n]
    uint16_t* __restrict__ ctx) {     // [b,n,D] bf16
  __shared__ uint16_t Plds[32 * 32];  // P tile bounce: C-layout -> A-layout

  const int lane = threadIdx.x;
  const int lr = lane & 15, hf = lane >> 4;
  const int qb = blockIdx.x & (MN / 32 - 1);
  const int bh = blockIdx.x >> 6;     // N/32 = 64 query blocks per (b,h)
  const size_t baseQK = (size_t)bh * MN * MDK;
  const size_t baseV  = (size_t)bh * MDK * MN;

  Frag16 qa[2][2];                     // 32 queries -> 2 row blocks x 2 dk
#pragma unroll
  for (int r = 0; r < 2; ++r) {
    const uint16_t* qrow =
        Qh + baseQK + (size_t)(qb * 32 + r * 16 + lr) * MDK;
#pragma unroll
    for (int d = 0; d < 2; ++d) {
      qa[r][d].q[0] = *(const u32x4*)(qrow + d * 32 + hf * 16);
      qa[r][d].q[1] = *(const u32x4*)(qrow + d * 32 + hf * 16 + 8);
    }
  }

  v8f o[2][4] = {{{}, {}, {}, {}}, {{}, {}, {}, {}}};
  float mrow[2][8], lrow[2][8];        // stats for row r*16 + v + 8*hf
#pragma unroll
  for (int r = 0; r < 2; ++r)
#pragma unroll
    for (int v = 0; v < 8; ++v) { mrow[r][v] = -1e30f; lrow[r][v] = 0.f; }

#pragma unroll 1   // keep live set < 256 VGPRs; no cross-iteration unroll
  for (int kc = 0; kc < MN; kc += 32) {
    // ---- S = Q K^T for 32 keys x 32 queries (K frag reused 2x) ----
    v8f s[2][2] = {{{}, {}}, {{}, {}}};
#pragma unroll
    for (int t = 0; t < 2; ++t) {
      const uint16_t* krow =
          Kh + baseQK + (size_t)(kc + t * 16 + lr) * MDK + hf * 16;
      __builtin_prefetch(krow + 32 * MDK, 0, 0);  // next chunk's key rows
#pragma unroll
      for (int d = 0; d < 2; ++d) {
        Frag16 kb;                     // B-frag: lane=key col, contiguous dk
        kb.q[0] = *(const u32x4*)(krow + d * 32);
        kb.q[1] = *(const u32x4*)(krow + d * 32 + 8);
#pragma unroll
        for (int r = 0; r < 2; ++r)
          s[r][t] = __builtin_amdgcn_wmma_f32_16x16x32_bf16(
              false, qa[r][d].v, false, kb.v, (short)0, s[r][t], false, false);
      }
    }

    // ---- online softmax in base-2 (rows reduced across 16 lanes) ----
    float alpha[2][8];
#pragma unroll
    for (int r = 0; r < 2; ++r) {
#pragma unroll
      for (int v = 0; v < 8; ++v) {
        float cm = fmaxf(s[r][0][v], s[r][1][v]);
#pragma unroll
        for (int off = 8; off; off >>= 1)
          cm = fmaxf(cm, __shfl_xor(cm, off, 32));
        const float mnew = fmaxf(mrow[r][v], cm);
        alpha[r][v] = exp2f(mrow[r][v] - mnew);
        const float p0 = exp2f(s[r][0][v] - mnew);
        const float p1 = exp2f(s[r][1][v] - mnew);
        s[r][0][v] = p0;
        s[r][1][v] = p1;
        float ps = p0 + p1;
#pragma unroll
        for (int off = 8; off; off >>= 1) ps += __shfl_xor(ps, off, 32);
        lrow[r][v] = lrow[r][v] * alpha[r][v] + ps;
        mrow[r][v] = mnew;
      }
#pragma unroll
      for (int t = 0; t < 4; ++t)
#pragma unroll
        for (int v = 0; v < 8; ++v) o[r][t][v] *= alpha[r][v];
    }

    // ---- P: C-layout -> LDS row-major [32q x 32k] -> A-frags ----
    __syncthreads();
#pragma unroll
    for (int r = 0; r < 2; ++r)
#pragma unroll
      for (int t = 0; t < 2; ++t)
#pragma unroll
        for (int v = 0; v < 8; ++v)
          Plds[(r * 16 + v + 8 * hf) * 32 + t * 16 + lr] = f2bf(s[r][t][v]);
    __syncthreads();

    Frag16 pf[2];
#pragma unroll
    for (int r = 0; r < 2; ++r) {
      const uint16_t* pr = &Plds[(r * 16 + lr) * 32 + hf * 16];
      pf[r].q[0] = *(const u32x4*)(pr);
      pf[r].q[1] = *(const u32x4*)(pr + 8);
    }

    // ---- O += P * V  (V frag reused 2x; B-frag contiguous keys) ----
#pragma unroll
    for (int t = 0; t < 4; ++t) {
      const uint16_t* vrow =
          Vt + baseV + (size_t)(t * 16 + lr) * MN + kc + hf * 16;
      __builtin_prefetch(vrow + 32, 0, 0);        // next chunk's keys
      Frag16 vb;
      vb.q[0] = *(const u32x4*)(vrow);
      vb.q[1] = *(const u32x4*)(vrow + 8);
#pragma unroll
      for (int r = 0; r < 2; ++r)
        o[r][t] = __builtin_amdgcn_wmma_f32_16x16x32_bf16(
            false, pf[r].v, false, vb.v, (short)0, o[r][t], false, false);
    }
  }

  // ---- finalize: O /= l, store ctx[b,n,h*64+dk] as bf16 ----
  const int b = bh / MH, h = bh % MH;
#pragma unroll
  for (int r = 0; r < 2; ++r)
#pragma unroll
    for (int v = 0; v < 8; ++v) {
      const float inv = 1.0f / lrow[r][v];
      const int qi = qb * 32 + r * 16 + v + 8 * hf;
#pragma unroll
      for (int t = 0; t < 4; ++t)
        ctx[(size_t)(b * MN + qi) * MD + h * MDK + t * 16 + lr] =
            f2bf(o[r][t][v] * inv);
    }
}

// ---------------------------------------------------------------------------
extern "C" void kernel_launch(void* const* d_in, const int* in_sizes, int n_in,
                              void* d_out, int out_size, void* d_ws,
                              size_t ws_size, hipStream_t stream) {
  (void)in_sizes; (void)n_in; (void)out_size; (void)ws_size;
  const float* q  = (const float*)d_in[0];
  const float* k  = (const float*)d_in[1];
  const float* v  = (const float*)d_in[2];
  const float* Wq = (const float*)d_in[3];
  const float* bq = (const float*)d_in[4];
  const float* Wk = (const float*)d_in[5];
  const float* bk = (const float*)d_in[6];
  const float* Wv = (const float*)d_in[7];
  const float* bv = (const float*)d_in[8];
  const float* Wo = (const float*)d_in[9];
  const float* bo = (const float*)d_in[10];
  float* out = (float*)d_out;

  // workspace carve-up (bf16 halves), 64 MB total, all 256B aligned
  const size_t SZ_BND = (size_t)MB * MN * MD;  // 4,194,304 elems
  const size_t SZ_W   = (size_t)MD * MD;       // 1,048,576 elems
  uint16_t* qb16 = (uint16_t*)d_ws;
  uint16_t* kb16 = qb16 + SZ_BND;
  uint16_t* vb16 = kb16 + SZ_BND;
  uint16_t* Wq16 = vb16 + SZ_BND;
  uint16_t* Wk16 = Wq16 + SZ_W;
  uint16_t* Wv16 = Wk16 + SZ_W;
  uint16_t* Wo16 = Wv16 + SZ_W;
  uint16_t* Qh   = Wo16 + SZ_W;
  uint16_t* Kh   = Qh + SZ_BND;
  uint16_t* Vt   = Kh + SZ_BND;
  uint16_t* ctx  = Vt + SZ_BND;

  // 1) fp32 -> bf16 for activations and weights
  cvt_f32_bf16<<<(unsigned)((SZ_BND + 255) / 256), 256, 0, stream>>>(q, qb16, (int)SZ_BND);
  cvt_f32_bf16<<<(unsigned)((SZ_BND + 255) / 256), 256, 0, stream>>>(k, kb16, (int)SZ_BND);
  cvt_f32_bf16<<<(unsigned)((SZ_BND + 255) / 256), 256, 0, stream>>>(v, vb16, (int)SZ_BND);
  cvt_f32_bf16<<<(unsigned)((SZ_W + 255) / 256), 256, 0, stream>>>(Wq, Wq16, (int)SZ_W);
  cvt_f32_bf16<<<(unsigned)((SZ_W + 255) / 256), 256, 0, stream>>>(Wk, Wk16, (int)SZ_W);
  cvt_f32_bf16<<<(unsigned)((SZ_W + 255) / 256), 256, 0, stream>>>(Wv, Wv16, (int)SZ_W);
  cvt_f32_bf16<<<(unsigned)((SZ_W + 255) / 256), 256, 0, stream>>>(Wo, Wo16, (int)SZ_W);

  // 2) QKV projections on WMMA (V written transposed for P*V B-frags;
  //    Q pre-scaled by DK^-0.5*log2(e))
  dim3 pgrid(MD / 64, (MB * MN) / 128);
  proj_wmma<0><<<pgrid, 128, 0, stream>>>(qb16, Wq16, bq, Qh);
  proj_wmma<1><<<pgrid, 128, 0, stream>>>(kb16, Wk16, bk, Kh);
  proj_wmma<2><<<pgrid, 128, 0, stream>>>(vb16, Wv16, bv, Vt);

  // 3) flash attention, one wave per 32-query block per (b,h)
  attn_wmma<<<MB * MH * (MN / 32), 32, 0, stream>>>(Qh, Kh, Vt, ctx);

  // 4) output projection straight to fp32 d_out
  proj_wmma<3><<<pgrid, 128, 0, stream>>>(ctx, Wo16, bo, out);
}
